// ROIAlign_63264868270542
// MI455X (gfx1250) — compile-verified, matrix operand order checked by reading, not run on hardware
//
#include <hip/hip_runtime.h>
#include <hip/hip_bf16.h>

// ---------------------------------------------------------------------------
// ROI-Align on MI455X (gfx1250), wave32 + WMMA + async global->LDS DMA.
//
// out(16x16) = Ay(16x64) * G(64x64) * Bx(64x16) per (box, channel), where
// G is the compact gathered tap grid, Ay/Bx are block-diagonal bilinear
// weight matrices. Stage 1 computed transposed (U^T = G^T * Ay^T) so stage-2
// A-fragments come from stage-1 D tiles with in-lane f32->f16 packing only.
// Crop windows are double-buffered in LDS and filled with
// GLOBAL_LOAD_ASYNC_TO_LDS_B32 (ASYNCcnt), pipelined one channel ahead.
// NOTE: window pointers are always direct GEPs off the extern-shared base so
// the gathers stay ds_load (addrspace(3)), never FLAT.
// ---------------------------------------------------------------------------

#define BN 2
#define KN 128
#define CN 256
#define NBOX (BN * KN)
#define IMGF 512.0f

// workspace layout per box (dwords)
#define WS_STRIDE 272
#define WSO_LVL   0
#define WSO_WL    1
#define WSO_YB    2
#define WSO_XB    3
#define WSO_HN    4
#define WSO_WN    5
#define WSO_YOFF  8      // pre-multiplied by 64 (window row pitch in floats)
#define WSO_XOFF  72
#define WSO_WY    136
#define WSO_WX    200

typedef __attribute__((ext_vector_type(16))) _Float16 v16h;
typedef __attribute__((ext_vector_type(8)))  float    v8f;

// K index for component j (j = 2*vgpr + half) of a 16-bit A/B fragment of
// v_wmma_*_16x16x32 (ISA 7.12.2): lanes 0-15 vs 16-31 differ by +8.
__device__ __forceinline__ int kmap(int j, int hi) {
    return ((j & 8) << 1) | (hi ? 8 : 0) | (j & 7);
}

// ---------------------------------------------------------------------------
// Prep kernel: one thread per box. Computes FPN level, the torch-style
// two-stage bilinear taps (16 outputs x 4 taps per axis), compacts them to a
// 64x64 window-relative grid, folds the validity mask into the row weights.
// ---------------------------------------------------------------------------
__device__ void make_taps(int start, int crop_len, int psize, float wscale,
                          int scaleOff,
                          int* __restrict__ offOut, float* __restrict__ wOut,
                          int* baseOut, int* needOut) {
    int   idx[64];
    float wt[64];
    int mn = 0x7fffffff, mx = -1;
    float clf = (float)crop_len;
    float psf = (float)psize;
    for (int o = 0; o < 16; ++o) {
        float s  = fmaxf(((float)o + 0.5f) * clf / 16.0f - 0.5f, 0.0f);
        float f0 = floorf(s);
        float w1 = s - f0;
        int f0i = (int)f0;
        int f1i = min(f0i + 1, crop_len - 1);

        float ga  = fmaxf(((float)(start + f0i) + 0.5f) * psf / IMGF - 0.5f, 0.0f);
        float ga0 = floorf(ga);
        float va  = ga - ga0;
        int a0 = (int)ga0;
        int a1 = min(a0 + 1, psize - 1);

        float gb  = fmaxf(((float)(start + f1i) + 0.5f) * psf / IMGF - 0.5f, 0.0f);
        float gb0 = floorf(gb);
        float vb  = gb - gb0;
        int b0 = (int)gb0;
        int b1 = min(b0 + 1, psize - 1);

        idx[4*o+0] = a0; idx[4*o+1] = a1; idx[4*o+2] = b0; idx[4*o+3] = b1;
        wt[4*o+0] = (1.0f - w1) * (1.0f - va);
        wt[4*o+1] = (1.0f - w1) * va;
        wt[4*o+2] = w1 * (1.0f - vb);
        wt[4*o+3] = w1 * vb;
        mn = min(mn, min(min(a0, a1), min(b0, b1)));
        mx = max(mx, max(max(a0, a1), max(b0, b1)));
    }
    for (int i = 0; i < 64; ++i) {
        offOut[i] = min(max(idx[i] - mn, 0), 63) << scaleOff;
        wOut[i]   = wt[i] * wscale;
    }
    *baseOut = mn;
    *needOut = min(mx - mn + 1, 64);
}

__global__ void roi_prep_kernel(const float* __restrict__ bboxess,
                                const int* __restrict__ counts,
                                float* __restrict__ ws) {
    int box = blockIdx.x * blockDim.x + threadIdx.x;
    if (box >= NBOX) return;
    int b = box / KN;
    int k = box % KN;

    const float* bb = bboxess + (size_t)box * 4;
    float y1f = bb[0], x1f = bb[1], y2f = bb[2], x2f = bb[3];

    float area = (y2f - y1f) * (x2f - x1f);
    float lvlf = 6.0f + log2f(sqrtf(fmaxf(area, 1e-30f)) / IMGF);
    int lvl = (int)rintf(lvlf);              // round-half-even, same as jnp.round
    lvl = min(5, max(2, lvl)) - 2;           // 0..3 -> p2..p5
    int psize = 128 >> lvl;                  // 128, 64, 32, 16 (square maps)

    int bi0 = (int)rintf(y1f), bi1 = (int)rintf(x1f);
    int bi2 = (int)rintf(y2f), bi3 = (int)rintf(x2f);
    int y1 = min(max(bi0, 0), 511);
    int x1 = min(max(bi1, 0), 511);
    int y2 = min(max(bi2, y1 + 1), 512);
    int x2 = min(max(bi3, x1 + 1), 512);

    float valid = (k < counts[b]) ? 1.0f : 0.0f;

    float* wsf = ws + (size_t)box * WS_STRIDE;
    int*   wsi = (int*)wsf;

    int yb, hn, xb, wn;
    // validity folded into y weights only (product weight -> zero output)
    // y offsets pre-scaled by window row pitch (64 floats)
    make_taps(y1, y2 - y1, psize, valid, 6, wsi + WSO_YOFF, wsf + WSO_WY, &yb, &hn);
    make_taps(x1, x2 - x1, psize, 1.0f,  0, wsi + WSO_XOFF, wsf + WSO_WX, &xb, &wn);

    wsi[WSO_LVL] = lvl;
    wsi[WSO_WL]  = psize;
    wsi[WSO_YB]  = yb;
    wsi[WSO_XB]  = xb;
    wsi[WSO_HN]  = hn;
    wsi[WSO_WN]  = wn;
}

// ---------------------------------------------------------------------------
// Main kernel: grid = (256 boxes, 16 channel-blocks), 128 threads = 4 waves.
// Each wave: 2 x 16 KB LDS crop windows (double buffer), 4 channels,
// 10 WMMAs per channel, async-DMA pipelined one channel ahead.
// ---------------------------------------------------------------------------
__global__ __launch_bounds__(128) void roi_wmma_kernel(
    const float* __restrict__ p2, const float* __restrict__ p3,
    const float* __restrict__ p4, const float* __restrict__ p5,
    const float* __restrict__ ws, float* __restrict__ out) {

    extern __shared__ char smemRaw[];
    int*   s_yoff = (int*)smemRaw;          // [64] pre-scaled *64
    int*   s_xoff = s_yoff + 64;            // [64]
    float* s_wy   = (float*)(s_xoff + 64);  // [64]
    float* s_wx   = s_wy + 64;              // [64]
    float* s_win  = s_wx + 64;              // [4 waves][2 bufs][64*64]

    const int box  = blockIdx.x;            // 0..255
    const int tid  = threadIdx.x;           // 0..127
    const int lane = tid & 31;
    const int wave = tid >> 5;              // 0..3
    const int hi   = (lane >= 16) ? 1 : 0;
    const int n    = lane & 15;

    const float* wsf = ws + (size_t)box * WS_STRIDE;
    const int*   wsi = (const int*)wsf;

    if (tid < 64) {
        s_yoff[tid] = wsi[WSO_YOFF + tid];
        s_xoff[tid] = wsi[WSO_XOFF + tid];
    } else {
        int t = tid - 64;
        s_wy[t] = wsf[WSO_WY + t];
        s_wx[t] = wsf[WSO_WX + t];
    }
    const int lvl   = wsi[WSO_LVL];
    const int Wl    = wsi[WSO_WL];
    const int ybase = wsi[WSO_YB];
    const int xbase = wsi[WSO_XB];
    const int hneed = wsi[WSO_HN];
    const int wneed = wsi[WSO_WN];

    const float* P = (lvl == 0) ? p2 : (lvl == 1) ? p3 : (lvl == 2) ? p4 : p5;
    const int plane = Wl * Wl;
    const int b     = box >> 7;             // K = 128
    const int slot  = blockIdx.y * 4 + wave;

    // LDS byte address of this wave's window pair (generic->LDS truncation,
    // done ONCE; all ds accesses below stay direct GEPs off s_win).
    const unsigned ldsWave = (unsigned)(uintptr_t)(s_win + wave * 2 * 4096);

    // Issue async global->LDS DMA for one channel's crop window (ASYNCcnt).
    auto issue_window = [&](int i) {
        const int c = slot * 4 + i;
        const float* Pc = P + ((size_t)(b * CN + c)) * plane;
        const unsigned ldsBase = ldsWave + (unsigned)(i & 1) * 16384u;
        for (int r = 0; r < hneed; ++r) {
            unsigned long long gbase =
                (unsigned long long)(uintptr_t)(Pc + (size_t)(ybase + r) * Wl + xbase);
            for (int col = lane; col < wneed; col += 32) {
                unsigned voff   = (unsigned)(col << 2);
                unsigned ldsDst = ldsBase + (unsigned)((r * 64 + col) << 2);
                asm volatile("global_load_async_to_lds_b32 %0, %1, %2"
                             :: "v"(ldsDst), "v"(voff), "s"(gbase)
                             : "memory");
            }
        }
    };

    // Prologue DMA overlaps the meta/fragment setup below.
    issue_window(0);
    __syncthreads();

    // Block-diagonal weight fragments, shared by all channels of this wave.
    // Stage-1 B = Ay^T  (K = row-tap ra, N = oh):   (ra>>2 == oh) ? wy[ra] : 0
    // Stage-2 B = Bx    (K = col-tap cb, N = ow):   (cb>>2 == ow) ? wx[cb] : 0
    v16h fragAyT[2], fragBx[2];
#pragma unroll
    for (int kc = 0; kc < 2; ++kc) {
#pragma unroll
        for (int j = 0; j < 16; ++j) {
            int kk = 32 * kc + kmap(j, hi);
            fragAyT[kc][j] = (_Float16)(((kk >> 2) == n) ? s_wy[kk] : 0.0f);
            fragBx[kc][j]  = (_Float16)(((kk >> 2) == n) ? s_wx[kk] : 0.0f);
        }
    }

    // Per-lane absolute column offsets for the 4 N-tiles of G^T.
    int xoffT[4];
#pragma unroll
    for (int t = 0; t < 4; ++t) xoffT[t] = s_xoff[16 * t + n];

#pragma unroll 1
    for (int i = 0; i < 4; ++i) {
        const int c = slot * 4 + i;
        // Direct GEP off extern-shared base => ds_load gathers (no FLAT).
        const float* win = s_win + (wave * 2 + (i & 1)) * 4096;

        // ---- wait for this channel's DMA, then gather all A fragments -----
        asm volatile("s_wait_asynccnt 0" ::: "memory");

        v16h afr[8];
#pragma unroll
        for (int t = 0; t < 4; ++t) {
#pragma unroll
            for (int kc = 0; kc < 2; ++kc) {
                v16h a = {};
#pragma unroll
                for (int j = 0; j < 16; ++j) {
                    int ra = 32 * kc + kmap(j, hi);   // yoff pre-scaled *64
                    a[j] = (_Float16)win[s_yoff[ra] + xoffT[t]];
                }
                afr[t * 2 + kc] = a;
            }
        }

        // ---- kick off next channel's DMA; overlaps all WMMA + stores ------
        if (i < 3) issue_window(i + 1);

        // ---- stage 1: U^T tiles = G^T * Ay^T  (8x wmma f16 K=32) ----------
        v8f accT[4];
#pragma unroll
        for (int t = 0; t < 4; ++t) {
            v8f acc = {};
            acc = __builtin_amdgcn_wmma_f32_16x16x32_f16(
                false, afr[t * 2 + 0], false, fragAyT[0], (short)0, acc, false, false);
            acc = __builtin_amdgcn_wmma_f32_16x16x32_f16(
                false, afr[t * 2 + 1], false, fragAyT[1], (short)0, acc, false, false);
            accT[t] = acc;
        }

        // ---- stage 2: out = U * Bx  (2x wmma f16 K=32) --------------------
        // A2_kc[j] = (f16) accT[2*kc + (j>>3)][j&7]  -- pure in-lane repack.
        v8f o = {};
#pragma unroll
        for (int kc = 0; kc < 2; ++kc) {
            v16h a2 = {};
#pragma unroll
            for (int j = 0; j < 16; ++j)
                a2[j] = (_Float16)accT[2 * kc + (j >> 3)][j & 7];
            o = __builtin_amdgcn_wmma_f32_16x16x32_f16(
                false, a2, false, fragBx[kc], (short)0, o, false, false);
        }

        // ---- write D tile: lane->(M = r + hi*8, N = lane&15) --------------
        float* dst = out + (((size_t)box * CN + c) << 8);  // * 256
        const int M0 = hi ? 8 : 0;
#pragma unroll
        for (int r = 0; r < 8; ++r)
            dst[(r + M0) * 16 + n] = o[r];
    }
}

// ---------------------------------------------------------------------------
extern "C" void kernel_launch(void* const* d_in, const int* in_sizes, int n_in,
                              void* d_out, int out_size, void* d_ws, size_t ws_size,
                              hipStream_t stream) {
    (void)in_sizes; (void)n_in; (void)out_size; (void)ws_size;
    const float* bboxess = (const float*)d_in[0];
    const int*   counts  = (const int*)d_in[1];
    const float* p2      = (const float*)d_in[2];
    const float* p3      = (const float*)d_in[3];
    const float* p4      = (const float*)d_in[4];
    const float* p5      = (const float*)d_in[5];
    // d_in[6] = p6 (never selected: level clamped to [2,5])
    float* out = (float*)d_out;
    float* ws  = (float*)d_ws;

    roi_prep_kernel<<<1, NBOX, 0, stream>>>(bboxess, counts, ws);

    dim3 grid(NBOX, 16);                     // 256 boxes x 16 channel-blocks
    size_t shmem = (size_t)(64 * 2) * sizeof(int)
                 + (size_t)(64 * 2) * sizeof(float)
                 + (size_t)4 * 2 * 64 * 64 * sizeof(float); // ~132 KB / workgroup
    roi_wmma_kernel<<<grid, 128, shmem, stream>>>(p2, p3, p4, p5, ws, out);
}
